// InternImageBlock_up_48747878810046
// MI455X (gfx1250) — compile-verified, matrix operand order checked by reading, not run on hardware
//
#include <hip/hip_runtime.h>

// ---------------------------------------------------------------- constants
#define NPIX   8192      // B*H*W
#define C_     256
#define G_     16
#define N2_    32768     // B*128*128
#define C2_    128

typedef __attribute__((ext_vector_type(16))) __bf16 v16bf;
typedef __attribute__((ext_vector_type(8)))  __bf16 v8bf;
typedef __attribute__((ext_vector_type(8)))  float  v8f;

#define DEV __device__ __forceinline__

DEV float gelu_f(float x) { return 0.5f * x * (1.0f + erff(x * 0.70710678118654752f)); }
DEV float sigmoid_f(float x) { return 1.0f / (1.0f + expf(-x)); }

union FragU { v16bf v; v8bf h[2]; };

// Load one wave's A or B fragment for v_wmma_f32_16x16x32_bf16 from global memory.
// rowptr = base of this lane's 16-element row (M or N = lane&15), row-major, K contiguous.
// Per ISA layout: lanes 0-15 hold K = {k0..k0+7, k0+16..k0+23}; lanes 16-31 get +8.
DEV v16bf load_frag(const __bf16* rowptr, int k0, int lane) {
  int koff = k0 + ((lane & 16) >> 1);   // +8 for upper half-wave
  FragU u;
  u.h[0] = *reinterpret_cast<const v8bf*>(rowptr + koff);
  u.h[1] = *reinterpret_cast<const v8bf*>(rowptr + koff + 16);
  return u.v;
}

// ---------------------------------------------------------------- conversions
__global__ void k_f32_to_bf16(const float* __restrict__ s, __bf16* __restrict__ d, long n) {
  long i = blockIdx.x * (long)blockDim.x + threadIdx.x;
  if (i < n) d[i] = (__bf16)s[i];
}

// w[Kd x Nc] row-major f32  ->  wt[Nc x Kd] row-major bf16 (transposed)
__global__ void k_wT_bf16(const float* __restrict__ w, __bf16* __restrict__ wt, int Kd, int Nc) {
  long i = blockIdx.x * (long)blockDim.x + threadIdx.x;
  if (i >= (long)Kd * Nc) return;
  int k = (int)(i / Nc), n = (int)(i % Nc);
  wt[(long)n * Kd + k] = (__bf16)w[i];
}

// up_w [3,3,256,128] -> per-tap transposed [tap][128][256] bf16
__global__ void k_upwT(const float* __restrict__ w, __bf16* __restrict__ wt) {
  int i = blockIdx.x * blockDim.x + threadIdx.x;   // 9*256*128
  if (i >= 9 * 256 * 128) return;
  int tap = i / (256 * 128), rem = i % (256 * 128);
  int ic = rem / 128, oc = rem % 128;
  wt[tap * (128 * 256) + oc * 256 + ic] = (__bf16)w[i];
}

// ---------------------------------------------------------------- WMMA GEMM
// out[Nr x Nc] = A[Nr x Kd](bf16) * BT[Nc x Kd]^T(bf16) + bias ; act: 1 = exact GELU.
// optional resid added after activation. block = (32,4): 4 waves.
// The 16-row A tile (shared by all 4 waves) is staged to LDS once with CDNA5
// async global->LDS copies (ASYNCcnt); A fragments then come from LDS while the
// per-wave B fragments stream from global. The K loop is software-pipelined so
// each WMMA overlaps with the next iteration's loads.
__global__ void k_gemm_bf16(const __bf16* __restrict__ A, const __bf16* __restrict__ BT,
                            const float* __restrict__ bias, const float* __restrict__ resid,
                            float* __restrict__ out, int Nr, int Kd, int Nc, int act) {
  __shared__ __align__(16) __bf16 sA[16 * 1024];   // up to Kd = 1024
  int lane = threadIdx.x;
  int ct = blockIdx.y * blockDim.y + threadIdx.y;
  int rt = blockIdx.x;
  bool active = (ct * 16 < Nc);                    // uniform per wave

  // ---- cooperative async staging of A tile: 16*Kd bf16 = 2*Kd 16B chunks ----
  {
    int tid = threadIdx.y * 32 + lane;             // 0..127
    const __bf16* gA = A + (long)rt * 16 * Kd;
    int nchunks = (16 * Kd) >> 3;                  // 8 bf16 (16B) per chunk
    for (int idx = tid; idx < nchunks; idx += 128) {
      unsigned lds_off = (unsigned)(uintptr_t)(&sA[idx * 8]);
      unsigned long long gaddr = (unsigned long long)(uintptr_t)(gA + idx * 8);
      asm volatile("global_load_async_to_lds_b128 %0, %1, off"
                   :: "v"(lds_off), "v"(gaddr) : "memory");
    }
    asm volatile("s_wait_asynccnt 0x0" ::: "memory");
  }
  __syncthreads();
  if (!active) return;

  const __bf16* brow = BT + (long)(ct * 16 + (lane & 15)) * Kd;
  int arow = (lane & 15) * Kd;
  int khalf = (lane & 16) >> 1;
  auto loadA = [&](int k0) {
    int koff = k0 + khalf;
    FragU u;
    u.h[0] = *reinterpret_cast<const v8bf*>(&sA[arow + koff]);
    u.h[1] = *reinterpret_cast<const v8bf*>(&sA[arow + koff + 16]);
    return u.v;
  };

  v8f acc = {};
  v16bf a0 = loadA(0);
  v16bf b0 = load_frag(brow, 0, lane);
  for (int k0 = 32; k0 < Kd; k0 += 32) {
    if (k0 + 32 < Kd) { __builtin_prefetch(brow + k0 + 32, 0, 3); }
    v16bf a1 = loadA(k0);
    v16bf b1 = load_frag(brow, k0, lane);
    acc = __builtin_amdgcn_wmma_f32_16x16x32_bf16(false, a0, false, b0, (short)0, acc,
                                                  false, false);
    a0 = a1; b0 = b1;
  }
  acc = __builtin_amdgcn_wmma_f32_16x16x32_bf16(false, a0, false, b0, (short)0, acc,
                                                false, false);

  int n = ct * 16 + (lane & 15);
  int mbase = rt * 16 + khalf;
  float bi = bias ? bias[n] : 0.0f;
#pragma unroll
  for (int r = 0; r < 8; ++r) {
    long m = mbase + r;
    float v = acc[r] + bi;
    if (act == 1) v = gelu_f(v);
    if (resid) v += resid[m * Nc + n];
    out[m * Nc + n] = v;
  }
}

// 3x3 conv 256->128 on zero-padded bf16 image [B,130,130,256] as 9 accumulated
// WMMA GEMMs, flattened into one 72-step software-pipelined loop.
__global__ void k_upconv_wmma(const __bf16* __restrict__ up, const __bf16* __restrict__ wt,
                              const float* __restrict__ bias, float* __restrict__ uc) {
  int lane = threadIdx.x;
  int ct = blockIdx.y * blockDim.y + threadIdx.y;
  if (ct * 16 >= C2_) return;
  int rt = blockIdx.x;                            // 0..2047
  int m = rt * 16 + (lane & 15);
  int b = m >> 14, y = (m >> 7) & 127, x = m & 127;
  const __bf16* bbase = wt + (long)(ct * 16 + (lane & 15)) * 256;

  auto loadA_s = [&](int s) {
    int tap = s >> 3, k0 = (s & 7) << 5;
    int dy = tap / 3, dx = tap % 3;
    const __bf16* arow = up + (((long)b * 130 + (y + dy)) * 130 + (x + dx)) * 256;
    return load_frag(arow, k0, lane);
  };
  auto loadB_s = [&](int s) {
    int tap = s >> 3, k0 = (s & 7) << 5;
    return load_frag(bbase + (long)tap * (128 * 256), k0, lane);
  };

  v8f acc = {};
  v16bf a0 = loadA_s(0);
  v16bf b0 = loadB_s(0);
  for (int s = 1; s < 72; ++s) {
    v16bf a1 = loadA_s(s);
    v16bf b1 = loadB_s(s);
    acc = __builtin_amdgcn_wmma_f32_16x16x32_bf16(false, a0, false, b0, (short)0, acc,
                                                  false, false);
    a0 = a1; b0 = b1;
  }
  acc = __builtin_amdgcn_wmma_f32_16x16x32_bf16(false, a0, false, b0, (short)0, acc,
                                                false, false);

  int n = ct * 16 + (lane & 15);
  int mb = rt * 16 + ((lane & 16) >> 1);
  float bi = bias[n];
#pragma unroll
  for (int r = 0; r < 8; ++r) uc[(long)(mb + r) * C2_ + n] = acc[r] + bi;
}

// ---------------------------------------------------------------- LayerNorm (+GELU)
// blockDim.x == C (256 or 128), dynamic LDS = C floats, 1 block per pixel.
__global__ void k_ln(const float* __restrict__ x, const float* __restrict__ g,
                     const float* __restrict__ b, float* __restrict__ o, int C, int act) {
  extern __shared__ float sh[];
  int p = blockIdx.x, c = threadIdx.x;
  float v = x[(long)p * C + c];
  sh[c] = v; __syncthreads();
  for (int s = C >> 1; s > 0; s >>= 1) { if (c < s) sh[c] += sh[c + s]; __syncthreads(); }
  float mean = sh[0] / (float)C;
  __syncthreads();
  float d = v - mean;
  sh[c] = d * d; __syncthreads();
  for (int s = C >> 1; s > 0; s >>= 1) { if (c < s) sh[c] += sh[c + s]; __syncthreads(); }
  float var = sh[0] / (float)C;
  float y = d * rsqrtf(var + 1e-6f) * g[c] + b[c];
  if (act) y = gelu_f(y);
  o[(long)p * C + c] = y;
}

// ---------------------------------------------------------------- depthwise conv (SAME)
__global__ void k_dwconv(const float* __restrict__ x, const float* __restrict__ w,
                         const float* __restrict__ bias, float* __restrict__ o, int ks) {
  long i = blockIdx.x * (long)blockDim.x + threadIdx.x;
  if (i >= (long)NPIX * C_) return;
  int c = (int)(i & (C_ - 1));
  long p = i >> 8;
  int b = (int)(p >> 12), y = (int)((p >> 6) & 63), xx = (int)(p & 63);
  int pad = ks >> 1;
  float acc = bias[c];
  for (int ky = 0; ky < ks; ++ky) {
    int yy = y + ky - pad; if ((unsigned)yy >= 64u) continue;
    for (int kx = 0; kx < ks; ++kx) {
      int xi = xx + kx - pad; if ((unsigned)xi >= 64u) continue;
      acc += x[(((long)b * 64 + yy) * 64 + xi) * C_ + c] * w[(ky * ks + kx) * C_ + c];
    }
  }
  o[i] = acc;
}

// ---------------------------------------------------------------- softmax over K=9
__global__ void k_softmax9(const float* __restrict__ in, float* __restrict__ out) {
  int i = blockIdx.x * blockDim.x + threadIdx.x;   // NPIX*G
  if (i >= NPIX * G_) return;
  int p = i / G_, g = i % G_;
  const float* s = in + (long)p * 144 + g * 9;
  float mx = s[0];
  for (int k = 1; k < 9; ++k) mx = fmaxf(mx, s[k]);
  float e[9], sum = 0.0f;
  for (int k = 0; k < 9; ++k) { e[k] = expf(s[k] - mx); sum += e[k]; }
  float inv = 1.0f / sum;
  float* d = out + (long)p * 144 + g * 9;
  for (int k = 0; k < 9; ++k) d[k] = e[k] * inv;
}

// ---------------------------------------------------------------- DCNv3 sampling
DEV void dcn_corner(float* acc, const float* __restrict__ xp, int b, int yy, int xx,
                    int g, float w) {
  int yi = yy - 1, xi = xx - 1;                    // padded -> image coords
  if ((unsigned)yi >= 64u || (unsigned)xi >= 64u) return;
  const float* s = xp + (((long)b * 64 + yi) * 64 + xi) * C_ + g * 16;
#pragma unroll
  for (int c = 0; c < 16; ++c) acc[c] += w * s[c];
}

__global__ void k_dcn_sample(const float* __restrict__ xp, const float* __restrict__ off,
                             const float* __restrict__ msk, float* __restrict__ samp) {
  int i = blockIdx.x * blockDim.x + threadIdx.x;   // NPIX*G
  if (i >= NPIX * G_) return;
  int p = i / G_, g = i % G_;
  int b = p >> 12, y = (p >> 6) & 63, x = p & 63;
  float acc[16];
#pragma unroll
  for (int c = 0; c < 16; ++c) acc[c] = 0.0f;
  const float* op = off + (long)p * 288 + g * 18;
  const float* mp = msk + (long)p * 144 + g * 9;
  for (int k = 0; k < 9; ++k) {
    float gx = (float)(k % 3) - 1.0f, gy = (float)(k / 3) - 1.0f;
    float px = (float)x + 1.0f + gx + op[k * 2 + 0];  // loc_x - 0.5 in padded coords
    float py = (float)y + 1.0f + gy + op[k * 2 + 1];
    float fx0 = floorf(px), fy0 = floorf(py);
    int x0 = (int)fx0, y0 = (int)fy0;
    float fx = px - fx0, fy = py - fy0;
    float m = mp[k];
    dcn_corner(acc, xp, b, y0,     x0,     g, (1.0f - fx) * (1.0f - fy) * m);
    dcn_corner(acc, xp, b, y0,     x0 + 1, g, fx * (1.0f - fy) * m);
    dcn_corner(acc, xp, b, y0 + 1, x0,     g, (1.0f - fx) * fy * m);
    dcn_corner(acc, xp, b, y0 + 1, x0 + 1, g, fx * fy * m);
  }
  float* d = samp + (long)p * C_ + g * 16;
#pragma unroll
  for (int c = 0; c < 16; ++c) d[c] = acc[c];
}

// ---------------------------------------------------------------- LEFFN residual
__global__ void k_leffn_res(float* __restrict__ t0, const float* __restrict__ t1,
                            const float* __restrict__ y, const float* __restrict__ gamma) {
  long i = blockIdx.x * (long)blockDim.x + threadIdx.x;
  if (i >= (long)NPIX * C_) return;
  int c = (int)(i & 255);
  t0[i] = t0[i] + t1[i] + gamma[c] * y[i];
}

// ---------------------------------------------------------------- CBAM
__global__ void k_chan_stats(const float* __restrict__ x, float* __restrict__ mean,
                             float* __restrict__ mx) {
  __shared__ float ssum[256], smax[256];
  int bc = blockIdx.x;                             // b*256 + c
  int b = bc >> 8, c = bc & 255;
  int t = threadIdx.x;
  float s = 0.0f, m = -3.4e38f;
  for (int p = t; p < 4096; p += 256) {
    float v = x[((long)b * 4096 + p) * C_ + c];
    s += v; m = fmaxf(m, v);
  }
  ssum[t] = s; smax[t] = m; __syncthreads();
  for (int st = 128; st > 0; st >>= 1) {
    if (t < st) { ssum[t] += ssum[t + st]; smax[t] = fmaxf(smax[t], smax[t + st]); }
    __syncthreads();
  }
  if (t == 0) { mean[bc] = ssum[0] * (1.0f / 4096.0f); mx[bc] = smax[0]; }
}

__global__ void k_cbam_fc(const float* __restrict__ mean, const float* __restrict__ mx,
                          const float* __restrict__ w1, const float* __restrict__ w2,
                          float* __restrict__ ca) {
  __shared__ float sm[256], sx[256], hm[16], hx[16];
  int b = blockIdx.x, t = threadIdx.x;
  sm[t] = mean[b * 256 + t]; sx[t] = mx[b * 256 + t]; __syncthreads();
  if (t < 16) {
    float a = 0.0f;
    for (int c = 0; c < 256; ++c) a += sm[c] * w1[c * 16 + t];
    hm[t] = fmaxf(a, 0.0f);
  } else if (t < 32) {
    int j = t - 16; float a = 0.0f;
    for (int c = 0; c < 256; ++c) a += sx[c] * w1[c * 16 + j];
    hx[j] = fmaxf(a, 0.0f);
  }
  __syncthreads();
  float o = 0.0f;
  for (int j = 0; j < 16; ++j) o += (hm[j] + hx[j]) * w2[j * 256 + t];
  ca[b * 256 + t] = sigmoid_f(o);
}

__global__ void k_ca_apply(const float* __restrict__ x, const float* __restrict__ ca,
                           float* __restrict__ xc, float* __restrict__ sm2) {
  __shared__ float ssum[256], smax[256];
  int p = blockIdx.x, c = threadIdx.x;
  int b = p >> 12;
  float v = x[(long)p * C_ + c] * ca[b * 256 + c];
  xc[(long)p * C_ + c] = v;
  ssum[c] = v; smax[c] = v; __syncthreads();
  for (int st = 128; st > 0; st >>= 1) {
    if (c < st) { ssum[c] += ssum[c + st]; smax[c] = fmaxf(smax[c], smax[c + st]); }
    __syncthreads();
  }
  if (c == 0) { sm2[p * 2 + 0] = ssum[0] * (1.0f / 256.0f); sm2[p * 2 + 1] = smax[0]; }
}

__global__ void k_sa_ln(const float* __restrict__ xc, const float* __restrict__ sm2,
                        const float* __restrict__ saw, const float* __restrict__ g,
                        const float* __restrict__ bb, float* __restrict__ t0) {
  __shared__ float sh[256];
  __shared__ float ssig;
  int p = blockIdx.x, c = threadIdx.x;
  int b = p >> 12, y = (p >> 6) & 63, x = p & 63;
  if (c == 0) {
    float s = 0.0f;
    for (int ky = 0; ky < 7; ++ky) {
      int yy = y + ky - 3; if ((unsigned)yy >= 64u) continue;
      for (int kx = 0; kx < 7; ++kx) {
        int xx = x + kx - 3; if ((unsigned)xx >= 64u) continue;
        const float* sp = sm2 + ((long)b * 4096 + yy * 64 + xx) * 2;
        const float* wp = saw + (ky * 7 + kx) * 2;
        s += sp[0] * wp[0] + sp[1] * wp[1];
      }
    }
    ssig = sigmoid_f(s);
  }
  __syncthreads();
  float v = xc[(long)p * C_ + c] * ssig;
  sh[c] = v; __syncthreads();
  for (int st = 128; st > 0; st >>= 1) { if (c < st) sh[c] += sh[c + st]; __syncthreads(); }
  float mean = sh[0] * (1.0f / 256.0f);
  __syncthreads();
  float d = v - mean;
  sh[c] = d * d; __syncthreads();
  for (int st = 128; st > 0; st >>= 1) { if (c < st) sh[c] += sh[c + st]; __syncthreads(); }
  float var = sh[0] * (1.0f / 256.0f);
  t0[(long)p * C_ + c] = d * rsqrtf(var + 1e-6f) * g[c] + bb[c];
}

// ---------------------------------------------------------------- upsample + pad
__global__ void k_upsample(const float* __restrict__ x, float* __restrict__ u) {
  long i = blockIdx.x * (long)blockDim.x + threadIdx.x;   // N2_*256
  if (i >= (long)N2_ * C_) return;
  int c = (int)(i & 255);
  long op = i >> 8;
  int b = (int)(op >> 14), yy = (int)((op >> 7) & 127), xx = (int)(op & 127);
  float sy = (yy + 0.5f) * 0.5f - 0.5f;
  float sx = (xx + 0.5f) * 0.5f - 0.5f;
  float fy0 = floorf(sy), fx0 = floorf(sx);
  int y0 = (int)fy0, x0 = (int)fx0;
  float fy = sy - fy0, fx = sx - fx0;
  int y0c = min(max(y0, 0), 63), y1c = min(max(y0 + 1, 0), 63);
  int x0c = min(max(x0, 0), 63), x1c = min(max(x0 + 1, 0), 63);
  const float* base = x + (long)b * 4096 * C_;
  float v00 = base[(y0c * 64 + x0c) * C_ + c];
  float v01 = base[(y0c * 64 + x1c) * C_ + c];
  float v10 = base[(y1c * 64 + x0c) * C_ + c];
  float v11 = base[(y1c * 64 + x1c) * C_ + c];
  u[i] = (v00 * (1.0f - fx) + v01 * fx) * (1.0f - fy) +
         (v10 * (1.0f - fx) + v11 * fx) * fy;
}

__global__ void k_pad_bf16(const float* __restrict__ u, __bf16* __restrict__ up) {
  long i = blockIdx.x * (long)blockDim.x + threadIdx.x;   // 2*130*130*256
  if (i >= (long)2 * 130 * 130 * 256) return;
  int c = (int)(i & 255);
  long pp = i >> 8;
  int b = (int)(pp / (130 * 130));
  int rem = (int)(pp % (130 * 130));
  int y = rem / 130, x = rem % 130;
  float v = 0.0f;
  if (y >= 1 && y <= 128 && x >= 1 && x <= 128)
    v = u[(((long)b * 128 + (y - 1)) * 128 + (x - 1)) * C_ + c];
  up[i] = (__bf16)v;
}

// ---------------------------------------------------------------- launch
extern "C" void kernel_launch(void* const* d_in, const int* in_sizes, int n_in,
                              void* d_out, int out_size, void* d_ws, size_t ws_size,
                              hipStream_t stream) {
  const float* x      = (const float*)d_in[0];
  const float* ln1_g  = (const float*)d_in[1];
  const float* ln1_b  = (const float*)d_in[2];
  const float* dw_w   = (const float*)d_in[3];
  const float* dw_b   = (const float*)d_in[4];
  const float* dwn_g  = (const float*)d_in[5];
  const float* dwn_b  = (const float*)d_in[6];
  const float* off_w  = (const float*)d_in[7];
  const float* off_b  = (const float*)d_in[8];
  const float* msk_w  = (const float*)d_in[9];
  const float* msk_b  = (const float*)d_in[10];
  const float* inp_w  = (const float*)d_in[11];
  const float* inp_b  = (const float*)d_in[12];
  const float* out_w  = (const float*)d_in[13];
  const float* out_b  = (const float*)d_in[14];
  const float* ln2_g  = (const float*)d_in[15];
  const float* ln2_b  = (const float*)d_in[16];
  const float* dc_w   = (const float*)d_in[17];
  const float* dc_b   = (const float*)d_in[18];
  const float* fn_g   = (const float*)d_in[19];
  const float* fn_b   = (const float*)d_in[20];
  const float* c1_w   = (const float*)d_in[21];
  const float* c1_b   = (const float*)d_in[22];
  const float* c2_w   = (const float*)d_in[23];
  const float* c2_b   = (const float*)d_in[24];
  const float* ffn_g  = (const float*)d_in[25];
  const float* ca_w1  = (const float*)d_in[26];
  const float* ca_w2  = (const float*)d_in[27];
  const float* sa_w   = (const float*)d_in[28];
  const float* cb_g   = (const float*)d_in[29];
  const float* cb_b   = (const float*)d_in[30];
  const float* nf_g   = (const float*)d_in[31];
  const float* nf_b   = (const float*)d_in[32];
  const float* up_w   = (const float*)d_in[33];
  const float* up_b   = (const float*)d_in[34];
  const float* up_g   = (const float*)d_in[35];
  const float* up_bt  = (const float*)d_in[36];
  float* outp = (float*)d_out;

  float* ws = (float*)d_ws;
  size_t cur = 0;
  auto alloc = [&](size_t nf) { float* p = ws + cur; cur += (nf + 15) & ~(size_t)15; return p; };
  float* t0    = alloc((size_t)NPIX * C_);
  float* t1    = alloc((size_t)NPIX * C_);
  float* xp    = alloc((size_t)NPIX * C_);
  float* fbr   = alloc((size_t)NPIX * C_);
  float* tmpA  = alloc((size_t)NPIX * C_);
  float* samp  = alloc((size_t)NPIX * C_);
  float* offb  = alloc((size_t)NPIX * 288);
  float* mskb  = alloc((size_t)NPIX * 144);
  float* h4    = alloc((size_t)NPIX * 1024);   // reused as upsample buffer `u`
  float* yffn  = alloc((size_t)NPIX * C_);
  float* uc    = alloc((size_t)N2_ * C2_);
  float* meanb = alloc(512);
  float* maxb  = alloc(512);
  float* cav   = alloc(512);
  float* smb   = alloc((size_t)NPIX * 2);
  __bf16* abf  = (__bf16*)alloc((size_t)NPIX * 1024 / 2);  // 8M bf16
  __bf16* wbf  = (__bf16*)alloc(147456);                    // up to 294912 bf16
  __bf16* upad = (__bf16*)alloc((size_t)2 * 130 * 130 * 256 / 2);
  float* u = h4;

  dim3 th256(256);
  dim3 wmma_blk(32, 4);
  auto g1 = [](long n) { return dim3((unsigned)((n + 255) / 256)); };

  // ---- CBAM ----
  k_chan_stats<<<512, th256, 0, stream>>>(x, meanb, maxb);
  k_cbam_fc<<<2, th256, 0, stream>>>(meanb, maxb, ca_w1, ca_w2, cav);
  k_ca_apply<<<NPIX, th256, 0, stream>>>(x, cav, tmpA, smb);
  k_sa_ln<<<NPIX, th256, 0, stream>>>(tmpA, smb, sa_w, cb_g, cb_b, t0);

  // ---- L = 2 blocks ----
  for (int i = 0; i < 2; ++i) {
    const float* L1g = ln1_g + i * 256; const float* L1b = ln1_b + i * 256;
    // t1 = LN(t0)
    k_ln<<<NPIX, th256, 1024, stream>>>(t0, L1g, L1b, t1, 256, 0);
    // xp = t1 @ inp_w + inp_b
    k_f32_to_bf16<<<g1((long)NPIX * 256), th256, 0, stream>>>(t1, abf, (long)NPIX * 256);
    k_wT_bf16<<<g1(65536), th256, 0, stream>>>(inp_w + i * 65536, wbf, 256, 256);
    k_gemm_bf16<<<dim3(512, 4), wmma_blk, 0, stream>>>(abf, wbf, inp_b + i * 256, nullptr,
                                                       xp, NPIX, 256, 256, 0);
    // f = gelu(LN(dw3(t1)))
    k_dwconv<<<g1((long)NPIX * 256), th256, 0, stream>>>(t1, dw_w + i * 9 * 256,
                                                         dw_b + i * 256, fbr, 3);
    k_ln<<<NPIX, th256, 1024, stream>>>(fbr, dwn_g + i * 256, dwn_b + i * 256, fbr, 256, 1);
    // off = f @ off_w + off_b ; msk = softmax(f @ msk_w + msk_b)
    k_f32_to_bf16<<<g1((long)NPIX * 256), th256, 0, stream>>>(fbr, abf, (long)NPIX * 256);
    k_wT_bf16<<<g1(256 * 288), th256, 0, stream>>>(off_w + i * 256 * 288, wbf, 256, 288);
    k_gemm_bf16<<<dim3(512, 5), wmma_blk, 0, stream>>>(abf, wbf, off_b + i * 288, nullptr,
                                                       offb, NPIX, 256, 288, 0);
    k_wT_bf16<<<g1(256 * 144), th256, 0, stream>>>(msk_w + i * 256 * 144, wbf, 256, 144);
    k_gemm_bf16<<<dim3(512, 3), wmma_blk, 0, stream>>>(abf, wbf, msk_b + i * 144, nullptr,
                                                       mskb, NPIX, 256, 144, 0);
    k_softmax9<<<g1(NPIX * G_), th256, 0, stream>>>(mskb, mskb);
    // deformable sampling, then output projection + residual into t0
    k_dcn_sample<<<g1(NPIX * G_), th256, 0, stream>>>(xp, offb, mskb, samp);
    k_f32_to_bf16<<<g1((long)NPIX * 256), th256, 0, stream>>>(samp, abf, (long)NPIX * 256);
    k_wT_bf16<<<g1(65536), th256, 0, stream>>>(out_w + i * 65536, wbf, 256, 256);
    k_gemm_bf16<<<dim3(512, 4), wmma_blk, 0, stream>>>(abf, wbf, out_b + i * 256, t0,
                                                       t0, NPIX, 256, 256, 0);
    // LEFFN: t1 = LN(t0); tmpA = LN(dw7(t1)); h4 = gelu(tmpA@c1); y = h4@c2; t0 += t1 + g*y
    k_ln<<<NPIX, th256, 1024, stream>>>(t0, ln2_g + i * 256, ln2_b + i * 256, t1, 256, 0);
    k_dwconv<<<g1((long)NPIX * 256), th256, 0, stream>>>(t1, dc_w + i * 49 * 256,
                                                         dc_b + i * 256, tmpA, 7);
    k_ln<<<NPIX, th256, 1024, stream>>>(tmpA, fn_g + i * 256, fn_b + i * 256, tmpA, 256, 0);
    k_f32_to_bf16<<<g1((long)NPIX * 256), th256, 0, stream>>>(tmpA, abf, (long)NPIX * 256);
    k_wT_bf16<<<g1(256 * 1024), th256, 0, stream>>>(c1_w + i * 256 * 1024, wbf, 256, 1024);
    k_gemm_bf16<<<dim3(512, 16), wmma_blk, 0, stream>>>(abf, wbf, c1_b + i * 1024, nullptr,
                                                        h4, NPIX, 256, 1024, 1);
    k_f32_to_bf16<<<g1((long)NPIX * 1024), th256, 0, stream>>>(h4, abf, (long)NPIX * 1024);
    k_wT_bf16<<<g1(1024 * 256), th256, 0, stream>>>(c2_w + i * 1024 * 256, wbf, 1024, 256);
    k_gemm_bf16<<<dim3(512, 4), wmma_blk, 0, stream>>>(abf, wbf, c2_b + i * 256, nullptr,
                                                       yffn, NPIX, 1024, 256, 0);
    k_leffn_res<<<g1((long)NPIX * 256), th256, 0, stream>>>(t0, t1, yffn, ffn_g + i * 256);
  }

  // ---- final LN + upsample + 3x3 conv (WMMA) + LN+GELU ----
  k_ln<<<NPIX, th256, 1024, stream>>>(t0, nf_g, nf_b, t1, 256, 0);
  k_upsample<<<g1((long)N2_ * C_), th256, 0, stream>>>(t1, u);
  k_pad_bf16<<<g1((long)2 * 130 * 130 * 256), th256, 0, stream>>>(u, upad);
  k_upwT<<<g1(9 * 256 * 128), th256, 0, stream>>>(up_w, wbf);
  k_upconv_wmma<<<dim3(2048, 2), wmma_blk, 0, stream>>>(upad, wbf, up_b, uc);
  k_ln<<<N2_, dim3(128), 512, stream>>>(uc, up_g, up_bt, outp, 128, 1);
}
